// MultiHeadAttention_62723702390946
// MI455X (gfx1250) — compile-verified, matrix operand order checked by reading, not run on hardware
//
#include <hip/hip_runtime.h>
#include <hip/hip_bf16.h>

// MHA forward: B=2, S=2048, D=1024, H=16, DK=64
// Pipeline: 3x f32->f16 WMMA projection GEMMs -> flash-attention WMMA kernel
// (TDM double-buffered K/V staging, DPP softmax reductions) -> WMMA output GEMM.

typedef __attribute__((ext_vector_type(16))) _Float16 v16h;
typedef __attribute__((ext_vector_type(8)))  _Float16 v8h;
typedef __attribute__((ext_vector_type(8)))  float    v8f;
typedef __attribute__((ext_vector_type(4)))  float    v4f;
typedef __attribute__((ext_vector_type(4)))  unsigned u32x4;
typedef __attribute__((ext_vector_type(8)))  unsigned u32x8;

#define BATCH   2
#define S_LEN   2048
#define DMODEL  1024
#define HEADS   16
#define DKH     64
#define MROWS   (BATCH * S_LEN)   // 4096

// padded LDS row strides (halves): +8 halves (16B) per row -> conflict-free
#define STR64   72                // for 64-half rows
#define STR32   40                // for 32-half rows

__device__ inline v8f wmma_f16(v16h a, v16h b, v8f c) {
    return __builtin_amdgcn_wmma_f32_16x16x32_f16(false, a, false, b, (short)0, c, false, false);
}

// 16x32 (rows x k) fragment from LDS. Per CDNA5 ISA: lane holds row;
// halves 0..7 = k[8*(lane>>4)..+7], halves 8..15 = k[16+8*(lane>>4)..+7].
__device__ inline v16h load_frag(const _Float16* base, int row, int stride, int lane) {
    const _Float16* p = base + row * stride + 8 * (lane >> 4);
    v8h lo = *(const v8h*)p;
    v8h hi = *(const v8h*)(p + 16);
    v16h r;
#pragma unroll
    for (int i = 0; i < 8; ++i) { r[i] = lo[i]; r[i + 8] = hi[i]; }
    return r;
}

// ---- DPP16 butterfly reductions across each 16-lane half-group -------------
__device__ inline float dpp_xor1(float v) {
    return __int_as_float(__builtin_amdgcn_update_dpp(0, __float_as_int(v), 0xB1, 0xf, 0xf, true));
}
__device__ inline float dpp_xor2(float v) {
    return __int_as_float(__builtin_amdgcn_update_dpp(0, __float_as_int(v), 0x4E, 0xf, 0xf, true));
}
__device__ inline float dpp_hmir(float v) {   // row_half_mirror (xor7 within 8)
    return __int_as_float(__builtin_amdgcn_update_dpp(0, __float_as_int(v), 0x141, 0xf, 0xf, true));
}
__device__ inline float dpp_mir(float v) {    // row_mirror (xor15 within 16)
    return __int_as_float(__builtin_amdgcn_update_dpp(0, __float_as_int(v), 0x140, 0xf, 0xf, true));
}
__device__ inline float rmax16(float v) {
    v = fmaxf(v, dpp_xor1(v));
    v = fmaxf(v, dpp_xor2(v));
    v = fmaxf(v, dpp_hmir(v));
    v = fmaxf(v, dpp_mir(v));
    return v;
}
__device__ inline float rsum16(float v) {
    v += dpp_xor1(v);
    v += dpp_xor2(v);
    v += dpp_hmir(v);
    v += dpp_mir(v);
    return v;
}

// ---- Tensor Data Mover helpers (CDNA5 ISA) -----------
__device__ inline unsigned lds_off(const void* p) {
    return (unsigned)(unsigned long long)(const __attribute__((address_space(3))) char*)p;
}

// 2D f16 tile load Global->LDS with LDS row padding.
// tile_d0 elements per row, tile_d1 rows (0 = 1D), d0_stride in elements,
// pad codes per D# spec: interval 0=2 DW,1=4,2=8,3=16,4=32; amount N -> N+1 DW.
__device__ inline void tdm_load_f16(unsigned ldsb, unsigned long long gaddr,
                                    unsigned tile_d0, unsigned tile_d1,
                                    unsigned tensor_d0, unsigned tensor_d1,
                                    unsigned long long d0_stride,
                                    unsigned pad_int, unsigned pad_amt) {
    u32x4 g0;
    g0.x = 1u;                                                    // count=1, user mode
    g0.y = ldsb;                                                  // lds_addr (bytes)
    g0.z = (unsigned)gaddr;                                       // global_addr[31:0]
    g0.w = ((unsigned)(gaddr >> 32) & 0x01FFFFFFu) | (2u << 30);  // addr[56:32] | type=2
    u32x8 g1;
    g1[0] = (1u << 16) | (1u << 20) | (pad_int << 22) | (pad_amt << 25); // 2B elems, pad_en
    g1[1] = (tensor_d0 & 0xFFFFu) << 16;
    g1[2] = (tensor_d0 >> 16) | ((tensor_d1 & 0xFFFFu) << 16);
    g1[3] = (tensor_d1 >> 16) | ((tile_d0 & 0xFFFFu) << 16);
    g1[4] = tile_d1 & 0xFFFFu;                                    // tile_dim1, tile_dim2=0
    g1[5] = (unsigned)d0_stride;
    g1[6] = (unsigned)(d0_stride >> 32) & 0xFFFFu;
    g1[7] = 0u;
    asm volatile("tensor_load_to_lds %0, %1" :: "s"(g0), "s"(g1) : "memory");
}

// ---------------------------------------------------------------------------
// GEMM: out = A(4096xK) @ W(NxK)^T + bias, K=N=1024.
// Block tile 128(M) x 64(N), 8 waves (4x2), each wave 32x32 = 2x2 WMMA tiles.
// MODE 0: f16 out, [B,H,S,DK]   MODE 1: f16 out, [B,H,DK,S]   MODE 2: f32 [B,S,D]
// ---------------------------------------------------------------------------
template <int MODE, bool A16>
__global__ __launch_bounds__(256) void gemm_xwT_kernel(const void* __restrict__ Ap,
                                                       const float* __restrict__ W,
                                                       const float* __restrict__ bias,
                                                       void* __restrict__ outp) {
    __shared__ __align__(16) _Float16 sA[128 * STR32];
    __shared__ __align__(16) _Float16 sB[64 * STR32];

    const int K = DMODEL;
    int t = threadIdx.x, wave = t >> 5, lane = t & 31;
    int bm = blockIdx.x;          // 0..31
    int bn = blockIdx.y;          // 0..15
    int wm = (wave >> 1) * 32;
    int wn = (wave & 1) * 32;

    v8f acc[2][2] = {};

    for (int k0 = 0; k0 < K; k0 += 32) {
        __syncthreads();
        // stage A tile 128x32 into LDS (f32->f16 convert unless already f16)
        {
            int row = t >> 1, c = (t & 1) * 16;
            _Float16* d = sA + row * STR32 + c;
            if (A16) {
                const _Float16* a = (const _Float16*)Ap + (size_t)(bm * 128 + row) * K + k0 + c;
                *(v8h*)d       = *(const v8h*)a;
                *(v8h*)(d + 8) = *(const v8h*)(a + 8);
            } else {
                const v4f* a4 = (const v4f*)((const float*)Ap + (size_t)(bm * 128 + row) * K + k0 + c);
                v4f x0 = a4[0], x1 = a4[1], x2 = a4[2], x3 = a4[3];
                v8h h0, h1;
#pragma unroll
                for (int i = 0; i < 4; ++i) {
                    h0[i] = (_Float16)x0[i]; h0[4 + i] = (_Float16)x1[i];
                    h1[i] = (_Float16)x2[i]; h1[4 + i] = (_Float16)x3[i];
                }
                *(v8h*)d = h0; *(v8h*)(d + 8) = h1;
            }
        }
        // stage B tile 64x32 (W is N-major K-contiguous = exactly the Bt layout)
        if (t < 128) {
            int row = t >> 1, c = (t & 1) * 16;
            const v4f* w4 = (const v4f*)(W + (size_t)(bn * 64 + row) * K + k0 + c);
            v4f x0 = w4[0], x1 = w4[1], x2 = w4[2], x3 = w4[3];
            v8h h0, h1;
#pragma unroll
            for (int i = 0; i < 4; ++i) {
                h0[i] = (_Float16)x0[i]; h0[4 + i] = (_Float16)x1[i];
                h1[i] = (_Float16)x2[i]; h1[4 + i] = (_Float16)x3[i];
            }
            _Float16* d = sB + row * STR32 + c;
            *(v8h*)d = h0; *(v8h*)(d + 8) = h1;
        }
        __syncthreads();

        v16h a0 = load_frag(sA + (size_t)wm * STR32,        lane & 15, STR32, lane);
        v16h a1 = load_frag(sA + (size_t)(wm + 16) * STR32, lane & 15, STR32, lane);
        v16h b0 = load_frag(sB + (size_t)wn * STR32,        lane & 15, STR32, lane);
        v16h b1 = load_frag(sB + (size_t)(wn + 16) * STR32, lane & 15, STR32, lane);

        acc[0][0] = wmma_f16(a0, b0, acc[0][0]);
        acc[0][1] = wmma_f16(a0, b1, acc[0][1]);
        acc[1][0] = wmma_f16(a1, b0, acc[1][0]);
        acc[1][1] = wmma_f16(a1, b1, acc[1][1]);
    }

    // C layout: lane holds N = lane&15; v8f element r is row M = r + 8*(lane>>4)
#pragma unroll
    for (int mi = 0; mi < 2; ++mi) {
#pragma unroll
        for (int ni = 0; ni < 2; ++ni) {
            int n = bn * 64 + wn + ni * 16 + (lane & 15);
            float bv = bias[n];
#pragma unroll
            for (int r = 0; r < 8; ++r) {
                int mrow = bm * 128 + wm + mi * 16 + 8 * (lane >> 4) + r;
                float v = acc[mi][ni][r] + bv;
                if (MODE == 2) {
                    ((float*)outp)[(size_t)mrow * DMODEL + n] = v;
                } else {
                    int b = mrow >> 11, s = mrow & (S_LEN - 1);
                    int h = n >> 6, dk = n & 63;
                    size_t idx;
                    if (MODE == 0)
                        idx = (((size_t)b * HEADS + h) * S_LEN + s) * DKH + dk;  // [B,H,S,DK]
                    else
                        idx = (((size_t)b * HEADS + h) * DKH + dk) * S_LEN + s;  // [B,H,DK,S]
                    ((_Float16*)outp)[idx] = (_Float16)v;
                }
            }
        }
    }
}

// ---------------------------------------------------------------------------
// Flash attention: grid = (S/128, B*H), 256 threads = 8 waves, wave owns 16 q rows.
// K/V tiles double-buffered via TDM tensor_load_to_lds; online softmax via DPP.
// ---------------------------------------------------------------------------
__global__ __launch_bounds__(256) void attn_kernel(const _Float16* __restrict__ q16,
                                                   const _Float16* __restrict__ k16,
                                                   const _Float16* __restrict__ vt16,
                                                   const int* __restrict__ mask,
                                                   _Float16* __restrict__ ctx16) {
    __shared__ __align__(16) _Float16 sQ[128 * STR64];       // 18 KB
    __shared__ __align__(16) _Float16 sK[2][32 * STR64];     // 2 x 4.5 KB
    __shared__ __align__(16) _Float16 sV[2][DKH * STR32];    // 2 x 5 KB (Vt: dk rows)
    __shared__ __align__(16) _Float16 sP[8 * 16 * STR32];    // 10 KB (per-wave P tiles)

    int t = threadIdx.x, wave = t >> 5, lane = t & 31;
    int qb = blockIdx.x;
    int bh = blockIdx.y;
    int b  = bh >> 4;
    int h  = bh & 15;

    const _Float16* qg = q16  + ((size_t)bh * S_LEN + qb * 128) * DKH;
    const _Float16* kg = k16  + (size_t)bh * S_LEN * DKH;
    const _Float16* vg = vt16 + (size_t)bh * DKH * S_LEN;
    const int* mk = mask + (size_t)b * S_LEN;

    const int NKB = S_LEN / 32;

    // prefetch K/V tile 0 via TDM (one wave issues; TENSORcnt is per-wave)
    if (wave == 0) {
        // K tile: 32x64 contiguous -> 1D copy of 2048 elems, pad 4 DW every 32 DW
        tdm_load_f16(lds_off(&sK[0][0]), (unsigned long long)kg,
                     2048u, 0u, 2048u, 1u, 2048u, 4u, 3u);
        // Vt tile: 64 rows x 32 elems, row stride S_LEN, pad 4 DW every 16 DW
        tdm_load_f16(lds_off(&sV[0][0]), (unsigned long long)vg,
                     32u, 64u, 32u, 64u, (unsigned long long)S_LEN, 3u, 3u);
    }

    // stage Q block (plain loads, padded rows)
    for (int i = t; i < 128 * DKH / 8; i += 256) {
        int row = i >> 3, c = (i & 7) * 8;
        *(v8h*)(sQ + row * STR64 + c) = *(const v8h*)(qg + row * DKH + c);
    }
    __syncthreads();

    const _Float16* qw = sQ + (size_t)(wave * 16) * STR64;
    v16h qa0 = load_frag(qw,      lane & 15, STR64, lane);   // k 0..31
    v16h qa1 = load_frag(qw + 32, lane & 15, STR64, lane);   // k 32..63

    _Float16* pbuf = sP + (size_t)(wave * 16) * STR32;

    v8f o0 = {}, o1 = {}, o2 = {}, o3 = {};
    float mrow[8], lrow[8];
#pragma unroll
    for (int r = 0; r < 8; ++r) { mrow[r] = -1e30f; lrow[r] = 0.0f; }

    for (int kb = 0; kb < NKB; ++kb) {
        int cur = kb & 1;
        // issue next tile's TDM, then wait for current tile (in-order per wave)
        if (wave == 0) {
            if (kb + 1 < NKB) {
                int nb = (kb + 1) & 1;
                tdm_load_f16(lds_off(&sK[nb][0]),
                             (unsigned long long)(kg + (size_t)(kb + 1) * 32 * DKH),
                             2048u, 0u, 2048u, 1u, 2048u, 4u, 3u);
                tdm_load_f16(lds_off(&sV[nb][0]),
                             (unsigned long long)(vg + (size_t)(kb + 1) * 32),
                             32u, 64u, 32u, 64u, (unsigned long long)S_LEN, 3u, 3u);
                __builtin_amdgcn_s_wait_tensorcnt(2);
            } else {
                __builtin_amdgcn_s_wait_tensorcnt(0);
            }
        }
        __syncthreads();   // current K/V tile visible to all waves

        const _Float16* kt = &sK[cur][0];
        const _Float16* vt = &sV[cur][0];

        // scores: two 16x16 C tiles over 32 key columns; DK=64 -> 2 WMMA each
        v8f s0 = {}, s1 = {};
        {
            v16h kf00 = load_frag(kt,      (lane & 15),      STR64, lane);
            v16h kf01 = load_frag(kt + 32, (lane & 15),      STR64, lane);
            v16h kf10 = load_frag(kt,      16 + (lane & 15), STR64, lane);
            v16h kf11 = load_frag(kt + 32, 16 + (lane & 15), STR64, lane);
            s0 = wmma_f16(qa0, kf00, s0);
            s0 = wmma_f16(qa1, kf01, s0);
            s1 = wmma_f16(qa0, kf10, s1);
            s1 = wmma_f16(qa1, kf11, s1);
        }

        int col0 = kb * 32 + (lane & 15);
        bool m0 = mk[col0] != 0;
        bool m1 = mk[col0 + 16] != 0;

#pragma unroll
        for (int r = 0; r < 8; ++r) {
            float f0 = m0 ? s0[r] * 0.125f : -1e9f;   // 1/sqrt(64)
            float f1 = m1 ? s1[r] * 0.125f : -1e9f;
            float bmax = rmax16(fmaxf(f0, f1));
            float mn = fmaxf(mrow[r], bmax);
            float sc = __expf(mrow[r] - mn);
            mrow[r] = mn;
            float p0 = __expf(f0 - mn);
            float p1 = __expf(f1 - mn);
            lrow[r] = lrow[r] * sc + rsum16(p0 + p1);
            o0[r] *= sc; o1[r] *= sc; o2[r] *= sc; o3[r] *= sc;
            int prow = r + 8 * (lane >> 4);
            pbuf[prow * STR32 + (lane & 15)]      = (_Float16)p0;
            pbuf[prow * STR32 + 16 + (lane & 15)] = (_Float16)p1;
        }
        // P is wave-private; LDS is in-order per wave -> wave-local fence only
        asm volatile("s_wait_dscnt 0x0" ::: "memory");

        // ctx += P(16x32) @ V(32x64): 4 WMMA (one per 16-wide dk tile)
        v16h pf  = load_frag(pbuf, lane & 15, STR32, lane);
        v16h vf0 = load_frag(vt + 0 * 16 * STR32, (lane & 15), STR32, lane);
        v16h vf1 = load_frag(vt + 1 * 16 * STR32, (lane & 15), STR32, lane);
        v16h vf2 = load_frag(vt + 2 * 16 * STR32, (lane & 15), STR32, lane);
        v16h vf3 = load_frag(vt + 3 * 16 * STR32, (lane & 15), STR32, lane);
        o0 = wmma_f16(pf, vf0, o0);
        o1 = wmma_f16(pf, vf1, o1);
        o2 = wmma_f16(pf, vf2, o2);
        o3 = wmma_f16(pf, vf3, o3);

        __syncthreads();   // all reads of cur done before TDM overwrites it
    }

    // normalize and store ctx as f16 in [B,S,D] (concat-of-heads) layout
#pragma unroll
    for (int r = 0; r < 8; ++r) {
        float inv = 1.0f / lrow[r];
        int qrow = qb * 128 + wave * 16 + 8 * (lane >> 4) + r;
        size_t base = ((size_t)b * S_LEN + qrow) * DMODEL + h * DKH;
        ctx16[base + 0 * 16 + (lane & 15)] = (_Float16)(o0[r] * inv);
        ctx16[base + 1 * 16 + (lane & 15)] = (_Float16)(o1[r] * inv);
        ctx16[base + 2 * 16 + (lane & 15)] = (_Float16)(o2[r] * inv);
        ctx16[base + 3 * 16 + (lane & 15)] = (_Float16)(o3[r] * inv);
    }
}

extern "C" void kernel_launch(void* const* d_in, const int* in_sizes, int n_in,
                              void* d_out, int out_size, void* d_ws, size_t ws_size,
                              hipStream_t stream) {
    (void)in_sizes; (void)n_in; (void)out_size; (void)ws_size;
    const float* preQ = (const float*)d_in[0];
    const float* preK = (const float*)d_in[1];
    const float* preV = (const float*)d_in[2];
    const float* Wq   = (const float*)d_in[3];
    const float* bq   = (const float*)d_in[4];
    const float* Wk   = (const float*)d_in[5];
    const float* bk   = (const float*)d_in[6];
    const float* Wv   = (const float*)d_in[7];
    const float* bv   = (const float*)d_in[8];
    const float* Wo   = (const float*)d_in[9];
    const float* bo   = (const float*)d_in[10];
    const int*   mask = (const int*)d_in[11];

    const size_t NELT = (size_t)BATCH * S_LEN * DMODEL;   // 4M halves each
    _Float16* ws    = (_Float16*)d_ws;
    _Float16* q16   = ws;
    _Float16* k16   = ws + NELT;
    _Float16* vt16  = ws + 2 * NELT;
    _Float16* ctx16 = ws + 3 * NELT;

    dim3 gg(MROWS / 128, DMODEL / 64);   // 32 x 16
    dim3 bb(256);
    gemm_xwT_kernel<0, false><<<gg, bb, 0, stream>>>(preQ, Wq, bq, q16);
    gemm_xwT_kernel<0, false><<<gg, bb, 0, stream>>>(preK, Wk, bk, k16);
    gemm_xwT_kernel<1, false><<<gg, bb, 0, stream>>>(preV, Wv, bv, vt16);

    attn_kernel<<<dim3(S_LEN / 128, BATCH * HEADS), bb, 0, stream>>>(q16, k16, vt16, mask, ctx16);

    gemm_xwT_kernel<2, true><<<gg, bb, 0, stream>>>(ctx16, Wo, bo, d_out);
}
